// model_benchmark_marabou_41686952575155
// MI455X (gfx1250) — matrix operand order repaired and checked by reading, not compiled, not run on hardware
//
#include <hip/hip_runtime.h>

typedef _Float16 h16;
typedef __attribute__((ext_vector_type(16))) _Float16 v16h;
typedef __attribute__((ext_vector_type(8)))  _Float16 v8h;
typedef __attribute__((ext_vector_type(8)))  float    v8f;

#define NWAVE 8
#define PW 5136            // halves per wave region
#define W_TOTAL 8704       // halves of staged weights
// per-wave region offsets (halves)
#define O_ARM 0
#define O_ACM 1024
#define O_BRM 1536
#define O_BCM 2560
#define O_MAT 3072
#define O_MRG 4096
#define O_GS  5120

struct Params {
  const float* inp;
  const float* W[16];
  const float* Bs[16];
  float* out;
  int nB;
};

__device__ __forceinline__ v8f wmma_f16(v16h a, v16h b, v8f c) {
  return __builtin_amdgcn_wmma_f32_16x16x32_f16(false, a, false, b, (short)0, c,
                                                false, false);
}

// A fragment (16x32 f16) from row-major (stride-32) LDS buffer.
// lane l: row m = l%16 (+16*mt), K halves koff..koff+15 contiguous, koff=(l<16?0:16)
__device__ __forceinline__ v16h load_a(const h16* rm, int lane, int mt) {
  int m = (lane & 15) + mt * 16;
  int koff = (lane & 16) ? 16 : 0;
  const h16* p = rm + m * 32 + koff;
  v8h lo = *(const v8h*)p;
  v8h hi = *(const v8h*)(p + 8);
  return __builtin_shufflevector(lo, hi, 0,1,2,3,4,5,6,7,8,9,10,11,12,13,14,15);
}

// B fragment (32x16 f16) from N-major (stride-32) LDS buffer.
// lane l: N = l%16 (+16*nt), K = koff..koff+15 contiguous, koff=(l<16?0:16)
__device__ __forceinline__ v16h load_b(const h16* nm, int lane, int nt) {
  int n = (lane & 15) + nt * 16;
  int koff = (lane & 16) ? 16 : 0;
  return *(const v16h*)(nm + n * 32 + koff);
}

// D (C-layout f32 16x16): lane l holds column n=l%16, rows r + (l<16?0:8), r=0..7
__device__ __forceinline__ void store_rm(h16* rm, int lane, int mt, int coff, v8f c) {
  int n = lane & 15;
  int mb = mt * 16 + ((lane & 16) ? 8 : 0);
#pragma unroll
  for (int r = 0; r < 8; r++) rm[(mb + r) * 32 + coff + n] = (h16)c[r];
}

__device__ __forceinline__ void store_cm(h16* cm, int lane, int mt, v8f c) {
  int n = lane & 15;
  int mb = mt * 16 + ((lane & 16) ? 8 : 0);
  v8h h;
#pragma unroll
  for (int r = 0; r < 8; r++) h[r] = (h16)c[r];
  *(v8h*)(cm + n * 32 + mb) = h;   // rows contiguous -> one ds_store_b128
}

__device__ __forceinline__ void relu_bias(v8f& c, float b) {
#pragma unroll
  for (int r = 0; r < 8; r++) c[r] = fmaxf(c[r] + b, 0.f);
}

// Z(20xDout<=16) = relu(X @ W^T + b) [+ residual]; src/dst row-major LDS
__device__ __forceinline__ void lin(const h16* src, h16* dst_rm, h16* dst_cm,
                                    v16h wf, float bval, int lane, const v8f* xres) {
#pragma unroll
  for (int mt = 0; mt < 2; mt++) {
    v16h a = load_a(src, lane, mt);
    v8f c = {};
    c = wmma_f16(a, wf, c);
    relu_bias(c, bval);
    if (xres) {
#pragma unroll
      for (int r = 0; r < 8; r++) c[r] += xres[mt][r];
    }
    store_rm(dst_rm, lane, mt, 0, c);
    if (dst_cm) store_cm(dst_cm, lane, mt, c);
  }
  __builtin_amdgcn_wave_barrier();
}

// D = MAT(20x20, staged A-layout source) @ Y (column-major LDS), no bias/relu
__device__ __forceinline__ void mm(const h16* mat, const h16* bcm, h16* dst_rm,
                                   h16* dst_cm, int coff, int lane) {
  v16h b = load_b(bcm, lane, 0);
#pragma unroll
  for (int mt = 0; mt < 2; mt++) {
    v16h a = load_a(mat, lane, mt);
    v8f c = {};
    c = wmma_f16(a, b, c);
    store_rm(dst_rm, lane, mt, coff, c);
    if (dst_cm) store_cm(dst_cm, lane, mt, c);
  }
  __builtin_amdgcn_wave_barrier();
}

__device__ __forceinline__ void stage20x20(h16* mat, const float* g, int lane) {
  for (int idx = lane; idx < 400; idx += 32)
    mat[(idx / 20) * 32 + (idx % 20)] = (h16)g[idx];
  __builtin_amdgcn_wave_barrier();
}

__global__ __launch_bounds__(256, 1)
void graphnet_wmma_kernel(Params P) {
  extern __shared__ h16 smem[];
  const int t = threadIdx.x;
  const int lane = t & 31;
  const int wave = t >> 5;

  // ---- stage weights to f16 B-layout (N-major, Kpad=32, zero-padded) ----
  constexpr int DIN[16]  = {5,16,8, 8,16,8, 13,16,8, 8,16,8, 29,32,16,8};
  constexpr int DOUT[16] = {16,8,8, 16,8,8, 16,8,8, 16,8,8, 32,16,8,1};
  constexpr int WOFF[16] = {0,512,1024,1536,2048,2560,3072,3584,4096,4608,5120,
                            5632,6144,7168,7680,8192};
#pragma unroll
  for (int L = 0; L < 16; L++) {
    const int slots = (L == 12) ? 32 : 16;
    h16* wb = smem + WOFF[L];
    const float* wg = P.W[L];
    for (int idx = t; idx < slots * 32; idx += 256) {
      int n = idx >> 5, k = idx & 31;
      float val = (n < DOUT[L] && k < DIN[L]) ? wg[n * DIN[L] + k] : 0.f;
      wb[idx] = (h16)val;
    }
  }
  __syncthreads();

  const int sample = blockIdx.x * NWAVE + wave;
  if (sample >= P.nB) return;                 // wave-uniform; after barrier
  const float* v = P.inp + (size_t)sample * 4300;
  h16* wr = smem + W_TOTAL + wave * PW;
  h16 *ARM = wr + O_ARM, *ACM = wr + O_ACM;
  h16 *BRM = wr + O_BRM, *BCM = wr + O_BCM;
  h16 *MAT = wr + O_MAT, *MRG = wr + O_MRG;
  float* GS = (float*)(wr + O_GS);

  { // zero wave region (padding must be zero)
    unsigned* z = (unsigned*)wr;
    for (int i = lane; i < PW / 2; i += 32) z[i] = 0u;
  }
  __builtin_amdgcn_wave_barrier();

  // node_inputs (20x5) -> MRG cols 0..4
  for (int idx = lane; idx < 100; idx += 32)
    MRG[(idx / 5) * 32 + (idx % 5)] = (h16)v[idx];
  __builtin_amdgcn_wave_barrier();

  const int n16 = lane & 15;
  float bh1 = P.Bs[0][n16], bh2 = (n16 < 8) ? P.Bs[1][n16] : 0.f,
        bh3 = (n16 < 8) ? P.Bs[2][n16] : 0.f;
  float bf1 = P.Bs[3][n16], bf2 = (n16 < 8) ? P.Bs[4][n16] : 0.f,
        bf3 = (n16 < 8) ? P.Bs[5][n16] : 0.f;
  float bd1 = P.Bs[6][n16], bd2 = (n16 < 8) ? P.Bs[7][n16] : 0.f,
        bd3 = (n16 < 8) ? P.Bs[8][n16] : 0.f;
  float bg1 = P.Bs[9][n16], bg2 = (n16 < 8) ? P.Bs[10][n16] : 0.f,
        bg3 = (n16 < 8) ? P.Bs[11][n16] : 0.f;
  float bfc1a = P.Bs[12][n16], bfc1b = P.Bs[12][n16 + 16];
  float bfc2 = P.Bs[13][n16], bfc3 = (n16 < 8) ? P.Bs[14][n16] : 0.f;
  float bfc4 = P.Bs[15][0];

  // ---- h chain; keep x = h3 output in registers (C layout) ----
  lin(MRG, ARM, nullptr, load_b(smem + WOFF[0], lane, 0), bh1, lane, nullptr);
  lin(ARM, BRM, nullptr, load_b(smem + WOFF[1], lane, 0), bh2, lane, nullptr);
  v8f xres[2];
  {
    v16h wf = load_b(smem + WOFF[2], lane, 0);
#pragma unroll
    for (int mt = 0; mt < 2; mt++) {
      v16h a = load_a(BRM, lane, mt);
      v8f c = {};
      c = wmma_f16(a, wf, c);
      relu_bias(c, bh3);
      store_rm(ARM, lane, mt, 0, c);
      xres[mt] = c;
    }
    __builtin_amdgcn_wave_barrier();
  }

  // ---- GCN loop: f-block weights hoisted ----
  v16h wf1 = load_b(smem + WOFF[3], lane, 0);
  v16h wf2 = load_b(smem + WOFF[4], lane, 0);
  v16h wf3 = load_b(smem + WOFF[5], lane, 0);
  h16 *p = ARM, *q = BRM, *pcm = ACM, *qcm = BCM;
  for (int it = 0; it < 8; it++) {
    stage20x20(MAT, v + 120 + it * 400, lane);
    if (it < 7) __builtin_prefetch(v + 120 + (it + 1) * 400, 0, 1);
    lin(p, q, nullptr, wf1, bf1, lane, nullptr);
    lin(q, p, nullptr, wf2, bf2, lane, nullptr);
    lin(p, q, qcm,    wf3, bf3, lane, nullptr);
    mm(MAT, qcm, p, nullptr, 0, lane);
    lin(p, q, nullptr, wf1, bf1, lane, nullptr);
    lin(q, p, nullptr, wf2, bf2, lane, nullptr);
    lin(p, q, nullptr, wf3, bf3, lane, xres);   // y = x + relu(f3(...))
    h16* tt = p; p = q; q = tt; tt = pcm; pcm = qcm; qcm = tt;
  }

  // gcn_output (in p) -> MRG cols 5..12
  for (int idx = lane; idx < 160; idx += 32)
    MRG[(idx >> 3) * 32 + 5 + (idx & 7)] = p[(idx >> 3) * 32 + (idx & 7)];
  __builtin_amdgcn_wave_barrier();

  // ---- d chain (input K=13 from MRG; cols 13..31 still zero here) ----
  lin(MRG, q, nullptr, load_b(smem + WOFF[6], lane, 0), bd1, lane, nullptr);
  lin(q, p, nullptr,   load_b(smem + WOFF[7], lane, 0), bd2, lane, nullptr);
  lin(p, q, qcm,       load_b(smem + WOFF[8], lane, 0), bd3, lane, nullptr);

  // dag_summary = summ_mats @ s
  stage20x20(MAT, v + 3480, lane);
  mm(MAT, qcm, p, pcm, 0, lane);

  // dag_extend = back_map @ dag_summary -> MRG cols 13..20 (21..28 overwritten later)
  stage20x20(MAT, v + 3900, lane);
  mm(MAT, pcm, MRG, nullptr, 13, lane);

  // ---- g chain on dag_summary (p) ----
  lin(p, q, nullptr, load_b(smem + WOFF[9], lane, 0),  bg1, lane, nullptr);
  lin(q, p, nullptr, load_b(smem + WOFF[10], lane, 0), bg2, lane, nullptr);
  lin(p, q, qcm,     load_b(smem + WOFF[11], lane, 0), bg3, lane, nullptr);

  // global_summary = running(1x20) @ s  (row 0 of one WMMA tile)
  for (int idx = lane; idx < 20; idx += 32) MAT[idx] = (h16)v[3880 + idx];
  __builtin_amdgcn_wave_barrier();
  {
    v16h a = load_a(MAT, lane, 0);
    v16h b = load_b(qcm, lane, 0);
    v8f c = {};
    c = wmma_f16(a, b, c);
    if (lane < 8) GS[lane] = c[0];         // m=0 row lives in VGPR0 of lanes 0..15
  }
  __builtin_amdgcn_wave_barrier();
  { // broadcast global_extend into MRG cols 21..28 (all 32 rows)
    float gv = GS[lane & 7];
    int m0 = lane >> 3;
#pragma unroll
    for (int j = 0; j < 8; j++)
      MRG[(m0 + 4 * j) * 32 + 21 + (lane & 7)] = (h16)gv;
  }
  __builtin_amdgcn_wave_barrier();

  // ---- fc chain: fc1 has Dout=32 -> two N tiles into q (32x32 RM) ----
  {
    v16h a0 = load_a(MRG, lane, 0), a1 = load_a(MRG, lane, 1);
#pragma unroll
    for (int nt = 0; nt < 2; nt++) {
      v16h wf = load_b(smem + WOFF[12], lane, nt);
      float bb = nt ? bfc1b : bfc1a;
      v8f c0 = {}, c1 = {};
      c0 = wmma_f16(a0, wf, c0); relu_bias(c0, bb);
      c1 = wmma_f16(a1, wf, c1); relu_bias(c1, bb);
      store_rm(q, lane, 0, nt * 16, c0);
      store_rm(q, lane, 1, nt * 16, c1);
    }
    __builtin_amdgcn_wave_barrier();
  }
  lin(q, p, nullptr, load_b(smem + WOFF[13], lane, 0), bfc2, lane, nullptr);
  lin(p, q, nullptr, load_b(smem + WOFF[14], lane, 0), bfc3, lane, nullptr);

  // fc4 (8->1), add bias + mask*1e4, write rows 0..19
  {
    v16h wf = load_b(smem + WOFF[15], lane, 0);
#pragma unroll
    for (int mt = 0; mt < 2; mt++) {
      v16h a = load_a(q, lane, mt);
      v8f c = {};
      c = wmma_f16(a, wf, c);
      if ((lane & 15) == 0) {
        int mb = mt * 16 + ((lane & 16) ? 8 : 0);
#pragma unroll
        for (int r = 0; r < 8; r++) {
          int m = mb + r;
          if (m < 20)
            P.out[(size_t)sample * 20 + m] = c[r] + bfc4 + v[100 + m] * 10000.f;
        }
      }
    }
  }
}

extern "C" void kernel_launch(void* const* d_in, const int* in_sizes, int n_in,
                              void* d_out, int out_size, void* d_ws, size_t ws_size,
                              hipStream_t stream) {
  (void)n_in; (void)d_ws; (void)ws_size;
  Params P;
  P.inp = (const float*)d_in[0];
  for (int i = 0; i < 16; i++) {
    P.W[i]  = (const float*)d_in[1 + 2 * i];
    P.Bs[i] = (const float*)d_in[2 + 2 * i];
  }
  P.out = (float*)d_out;
  P.nB  = in_sizes[0] / 4300;
  (void)out_size;
  const int blocks = (P.nB + NWAVE - 1) / NWAVE;
  const size_t shmem = (size_t)(W_TOTAL + NWAVE * PW) * sizeof(h16); // ~99.6 KB
  graphnet_wmma_kernel<<<blocks, 256, shmem, stream>>>(P);
}
